// WP_31078383354692
// MI455X (gfx1250) — matrix operand order repaired and checked by reading, not compiled, not run on hardware
//
#include <hip/hip_runtime.h>
#include <hip/hip_bf16.h>

typedef __attribute__((ext_vector_type(16))) _Float16 v16h;
typedef __attribute__((ext_vector_type(8)))  _Float16 v8h;
typedef __attribute__((ext_vector_type(8)))  float    v8f;

#define BATCH 32
#define SEQL  64
#define DIM   768
#define TWOD  1536

// ---------------------------------------------------------------------------
// Kernel 0: f32 -> f16 conversion (fc_w staging for WMMA B-matrix)
// ---------------------------------------------------------------------------
__global__ void cvt_f32_to_f16(const float* __restrict__ src,
                               _Float16* __restrict__ dst, int n) {
  int i = blockIdx.x * blockDim.x + threadIdx.x;
  if (i < n) dst[i] = (_Float16)src[i];
}

// ---------------------------------------------------------------------------
// Kernel 1: fused attention-pool per batch.
//   p1 = a1 . w ; p2 = a2 . w                (B,L)
//   logits[i*l2+j] = p1[1+i] - p2[1+j]  (valid), masked -> softmax
//   r1[i] = sum_j e, r2[j] = sum_i e   (LDS ds_add_f32 atomics)
//   pooled = (sum_i r1[i]*a1[1+i] - sum_j r2[j]*a2[1+j]) / Z  -> f16 ws
// ---------------------------------------------------------------------------
__global__ __launch_bounds__(256)
void pool_kernel(const float* __restrict__ a1, const float* __restrict__ a2,
                 const int* __restrict__ m1, const int* __restrict__ m2,
                 const float* __restrict__ we, _Float16* __restrict__ pooledh) {
  __shared__ float p1[SEQL], p2[SEQL], r1[SEQL], r2[SEQL], red[256];
  __shared__ float gmax, zsum;
  __shared__ int   sl1, sl2;

  const int b = blockIdx.x;
  const int tid = threadIdx.x;

  // content lengths (two waves in parallel, trivial work)
  if (tid == 0)  { int s = 0; for (int k = 0; k < SEQL; ++k) s += m1[b*SEQL+k]; sl1 = s - 2; }
  if (tid == 32) { int s = 0; for (int k = 0; k < SEQL; ++k) s += m2[b*SEQL+k]; sl2 = s - 2; }

  // --- p1 / p2: 2 arrays * 64 rows * 2 half-rows = 256 tasks ---
  {
    const int arr  = tid >> 7;           // 0 -> a1, 1 -> a2
    const int row  = (tid & 127) >> 1;   // 0..63
    const int half = tid & 1;            // 0..1
    const float* src = (arr ? a2 : a1) + (size_t)(b*SEQL + row)*DIM + half*(DIM/2);
    const float* wv  = we + half*(DIM/2);
    float s = 0.f;
    #pragma unroll 4
    for (int d = 0; d < DIM/2; ++d) s += src[d] * wv[d];
    red[tid] = s;
  }
  if (tid < SEQL) { r1[tid] = 0.f; r2[tid] = 0.f; }
  __syncthreads();
  if ((tid & 1) == 0) {
    const float v  = red[tid] + red[tid + 1];
    const int arr  = tid >> 7;
    const int row  = (tid & 127) >> 1;
    if (arr) p2[row] = v; else p1[row] = v;
  }
  __syncthreads();

  const int l1 = sl1, l2 = sl2, nvalid = l1 * l2;

  // --- pass 1: global max of masked logits over the L*L grid ---
  float lmax = -3.0e38f;
  for (int kk = tid; kk < SEQL*SEQL; kk += 256) {
    float mval = -1e7f;
    if (kk < nvalid) {
      const int i = kk / l2, j = kk - i * l2;
      const float w = p1[i + 1] - p2[j + 1];
      mval = (fabsf(w) < 1e-7f) ? -1e7f : w;
    }
    lmax = fmaxf(lmax, mval);
  }
  red[tid] = lmax; __syncthreads();
  for (int s = 128; s > 0; s >>= 1) {
    if (tid < s) red[tid] = fmaxf(red[tid], red[tid + s]);
    __syncthreads();
  }
  if (tid == 0) gmax = red[0];
  __syncthreads();
  const float mx = gmax;
  __syncthreads();

  // --- pass 2: Z = sum(exp) over ALL entries; r1/r2 over valid entries ---
  float lsum = 0.f;
  for (int kk = tid; kk < SEQL*SEQL; kk += 256) {
    float mval = -1e7f; int i = 0, j = 0;
    const bool valid = kk < nvalid;
    if (valid) {
      i = kk / l2; j = kk - i * l2;
      const float w = p1[i + 1] - p2[j + 1];
      mval = (fabsf(w) < 1e-7f) ? -1e7f : w;
    }
    const float e = __expf(mval - mx);
    lsum += e;
    if (valid) {                       // LDS float atomics -> ds_add_f32
      atomicAdd(&r1[i], e);
      atomicAdd(&r2[j], e);
    }
  }
  red[tid] = lsum; __syncthreads();
  for (int s = 128; s > 0; s >>= 1) {
    if (tid < s) red[tid] += red[tid + s];
    __syncthreads();
  }
  if (tid == 0) zsum = red[0];
  __syncthreads();
  const float invZ = 1.0f / zsum;

  // --- pooled: weighted row combination, emit f16 for the WMMA A-matrix ---
  for (int d = tid; d < DIM; d += 256) {
    float acc = 0.f;
    for (int i = 0; i < SEQL - 1; ++i) {   // rows 1..63; r[i]==0 beyond l
      const size_t ro = (size_t)(b*SEQL + i + 1)*DIM + d;
      acc += r1[i] * a1[ro];
      acc -= r2[i] * a2[ro];
    }
    pooledh[b*DIM + d] = (_Float16)(acc * invZ);
  }
}

// ---------------------------------------------------------------------------
// Kernel 2: out(32x1536) = tanh(pooled(32x768) @ fc_w^T + fc_b)
// via v_wmma_f32_16x16x32_f16. One wave per 16x16 output tile:
// 2 M-tiles * 96 N-tiles = 192 waves = 48 blocks * 4 waves. K = 768 = 24 steps.
// ---------------------------------------------------------------------------
__global__ __launch_bounds__(128)
void gemm_kernel(const _Float16* __restrict__ Aws,   // (32,768) pooled, row-major
                 const _Float16* __restrict__ Bws,   // (1536,768) fc_w, row-major (K contiguous)
                 const float* __restrict__ bias,
                 float* __restrict__ out) {
  const int lane  = threadIdx.x & 31;
  const int wave  = blockIdx.x * (blockDim.x >> 5) + (threadIdx.x >> 5); // 0..191
  const int mTile = wave & 1;
  const int nTile = wave >> 1;                 // 0..95

  const int mrow  = (mTile << 4) + (lane & 15);
  const int ncol  = (nTile << 4) + (lane & 15);
  const int aKoff = (lane < 16) ? 0 : 16;      // 16-bit A striping: K 0..15 vs 16..31
  const int bKoff = (lane < 16) ? 0 : 16;      // B striping: K 0..15 vs 16..31

  const _Float16* aRow = Aws + (size_t)mrow * DIM;
  const _Float16* bRow = Bws + (size_t)ncol * DIM;

  v8f acc = {};
  for (int ks = 0; ks < DIM; ks += 32) {
    __builtin_prefetch(bRow + ks + 128, 0, 0); // global_prefetch_b8, next tiles
    const v8h alo = *(const v8h*)(aRow + ks + aKoff);
    const v8h ahi = *(const v8h*)(aRow + ks + aKoff + 8);
    v16h a;
    #pragma unroll
    for (int t = 0; t < 8; ++t) { a[t] = alo[t]; a[t + 8] = ahi[t]; }
    const v16h bfrag = *(const v16h*)(bRow + ks + bKoff);
    acc = __builtin_amdgcn_wmma_f32_16x16x32_f16(
        /*neg_a=*/false, a, /*neg_b=*/false, bfrag,
        /*c_mod=*/(short)0, acc, /*reuse_a=*/false, /*reuse_b=*/false);
  }

  const float bn = bias[ncol];
  const int rbase = (mTile << 4) + ((lane < 16) ? 0 : 8);
  #pragma unroll
  for (int v = 0; v < 8; ++v) {
    out[(size_t)(rbase + v) * TWOD + ncol] = tanhf(acc[v] + bn);
  }
}

// ---------------------------------------------------------------------------
extern "C" void kernel_launch(void* const* d_in, const int* in_sizes, int n_in,
                              void* d_out, int out_size, void* d_ws, size_t ws_size,
                              hipStream_t stream) {
  (void)in_sizes; (void)n_in; (void)out_size; (void)ws_size;
  const float* a1  = (const float*)d_in[0];
  const float* a2  = (const float*)d_in[1];
  const int*   m1  = (const int*)d_in[2];
  const int*   m2  = (const int*)d_in[3];
  const float* we  = (const float*)d_in[4];
  const float* fcw = (const float*)d_in[5];
  const float* fcb = (const float*)d_in[6];
  float* out = (float*)d_out;

  _Float16* Aws = (_Float16*)d_ws;                       // 32*768 halves (48 KB)
  _Float16* Bws = (_Float16*)((char*)d_ws + (64 << 10)); // 1536*768 halves (~2.25 MB)

  const int nB = TWOD * DIM;
  cvt_f32_to_f16<<<(nB + 255) / 256, 256, 0, stream>>>(fcw, Bws, nB);
  pool_kernel<<<BATCH, 256, 0, stream>>>(a1, a2, m1, m2, we, Aws);
  gemm_kernel<<<48, 128, 0, stream>>>(Aws, Bws, fcb, out);
}
